// AttentionBlock_6708738916527
// MI455X (gfx1250) — matrix operand order repaired and checked by reading, not compiled, not run on hardware
//
#include <hip/hip_runtime.h>

typedef __attribute__((ext_vector_type(16))) _Float16 v16h;
typedef __attribute__((ext_vector_type(8)))  _Float16 v8h;
typedef __attribute__((ext_vector_type(2)))  _Float16 h2;
typedef __attribute__((ext_vector_type(8)))  float    v8f;
typedef __attribute__((ext_vector_type(4)))  float    v4f;

#define NTOK 4096   // 16*16*16 tokens

// ---- native exp2 (v_exp_f32) ----------------------------------------------
__device__ __forceinline__ float fexp2(float x) {
#if __has_builtin(__builtin_amdgcn_exp2f)
    return __builtin_amdgcn_exp2f(x);
#else
    return exp2f(x);
#endif
}
// ---- lane^16 exchange (lower<->upper half of wave32), VALU op --------------
__device__ __forceinline__ int xswap16i(int v) {
#if __has_builtin(__builtin_amdgcn_permlanex16)
    return __builtin_amdgcn_permlanex16(v, v, 0x76543210, (int)0xfedcba98,
                                        true, false);
#else
    return __shfl_xor(v, 16, 32);
#endif
}
__device__ __forceinline__ float xswap16f(float v) {
    return __builtin_bit_cast(float, xswap16i(__builtin_bit_cast(int, v)));
}
// ---- pack two f32 -> packed f16 dword --------------------------------------
__device__ __forceinline__ int pack2(float a, float b) {
#if __has_builtin(__builtin_amdgcn_cvt_pkrtz)
    return __builtin_bit_cast(int, __builtin_amdgcn_cvt_pkrtz(a, b));
#else
    h2 p; p[0] = (_Float16)a; p[1] = (_Float16)b;
    return __builtin_bit_cast(int, p);
#endif
}
// ---- f16 WMMA 16x16x32, f32 accumulate -------------------------------------
__device__ __forceinline__ v8f wmma16(v16h a, v16h b, v8f c) {
    return __builtin_amdgcn_wmma_f32_16x16x32_f16(false, a, false, b,
                                                  (short)0, c, false, false);
}
// ---- A-operand load: halves 0..15 -> k=16*hi+h ------------------------------
__device__ __forceinline__ v16h loadA(const _Float16* base, int hi) {
    v16h a;
    ((v8h*)&a)[0] = *(const v8h*)(base + 16 * hi);
    ((v8h*)&a)[1] = *(const v8h*)(base + 16 * hi + 8);
    return a;
}
// ---- build P^T B-operand from two C-layout tiles (one lane^16 swap) --------
__device__ __forceinline__ v16h buildPB(const v8f& t0, const v8f& t1, int hi) {
    int pk0[4], pk1[4];
    #pragma unroll
    for (int r = 0; r < 4; ++r) {
        pk0[r] = pack2(t0[2 * r], t0[2 * r + 1]);
        pk1[r] = pack2(t1[2 * r], t1[2 * r + 1]);
    }
    v16h pb;
    int* w = (int*)&pb;
    #pragma unroll
    for (int r = 0; r < 4; ++r) {
        int rcv = xswap16i(hi ? pk0[r] : pk1[r]);
        w[r]     = hi ? rcv : pk0[r];
        w[4 + r] = hi ? pk1[r] : rcv;
    }
    return pb;
}

// ---------------------------------------------------------------------------
// Kernel 1: QKV projection.
//   x: [B][64][4096] (channels-major).  qkv = x^T @ w_proj^T + b_proj.
// Outputs f16: Q [B][N][64] (pre-scaled by log2(e)/8 so exp == exp2),
//              K [B][N][64], Vt [B][64][N].
// ---------------------------------------------------------------------------
__global__ __launch_bounds__(256) void qkv_kernel(
    const float* __restrict__ x, const float* __restrict__ wproj,
    const float* __restrict__ bproj,
    _Float16* __restrict__ Qh, _Float16* __restrict__ Kh,
    _Float16* __restrict__ Vth)
{
    __shared__ float xs[64][65];
    const int blk   = blockIdx.x;        // 0..127  (2 b * 64 n-tiles)
    const int b     = blk >> 6;
    const int nbase = (blk & 63) << 6;
    const int tid   = threadIdx.x;
    const int nn    = tid & 63;
    const int cb    = tid >> 6;

    #pragma unroll
    for (int k = 0; k < 16; ++k) {
        int c = cb + k * 4;
        xs[c][nn] = x[(size_t)((b << 6) + c) * NTOK + nbase + nn];
    }
    __syncthreads();

    const float qscale = 0.125f * 1.44269504088896340736f;  // dk^-.5 * log2(e)
    for (int o = cb; o < 192; o += 4) {
        float acc = bproj[o];
        const float* wr = wproj + o * 64;   // uniform across the n-group
        #pragma unroll 8
        for (int c = 0; c < 64; ++c) acc += xs[c][nn] * wr[c];
        const int n = nbase + nn;
        if (o < 64) {
            Qh[(size_t)((b << 12) + n) * 64 + o] = (_Float16)(acc * qscale);
        } else if (o < 128) {
            Kh[(size_t)((b << 12) + n) * 64 + (o - 64)] = (_Float16)acc;
        } else {
            Vth[(size_t)((b << 6) + (o - 128)) * NTOK + n] = (_Float16)acc;
        }
    }
}

// ---------------------------------------------------------------------------
// Kernel 2: flash attention (transposed formulation), one wave per 32 queries
// (two 16-query groups A/B share every K/V tile load -> 16 WMMA per 256B).
//   S^T = K Q^T -> P^T = exp2(S^T) -> O^T = V^T P^T ; l = col-sums of P^T.
// Query index lives in the lane dimension => softmax stats are lane-local.
// ---------------------------------------------------------------------------
__global__ __launch_bounds__(64) void flash_kernel(
    const _Float16* __restrict__ Qh, const _Float16* __restrict__ Kh,
    const _Float16* __restrict__ Vth, float* __restrict__ Ows)
{
    const int ln = threadIdx.x & 31;
    const int lo = ln & 15;
    const int hi = ln >> 4;
    const int tp = blockIdx.x * 2 + (threadIdx.x >> 5);   // 0..255
    const int b  = tp >> 7;
    const int i0 = (tp & 127) << 5;                       // 32 queries

    const _Float16* Qb = Qh  + (size_t)(b << 12) * 64;
    const _Float16* Kb = Kh  + (size_t)(b << 12) * 64;
    const _Float16* Vb = Vth + (size_t)(b << 6) * NTOK;

    // Q as B-operands (lane = query column, halves k = d = 16*hi + h)
    v16h qbA0, qbA1, qbB0, qbB1;
    {
        const _Float16* qr = Qb + (size_t)(i0 + lo) * 64 + 16 * hi;
        qbA0 = *(const v16h*)(qr);
        qbA1 = *(const v16h*)(qr + 32);
        qr += 16 * 64;
        qbB0 = *(const v16h*)(qr);
        qbB1 = *(const v16h*)(qr + 32);
    }

    v8f otA[4] = {}, otB[4] = {};
    float lsumA = 0.f, lsumB = 0.f;
    const v8f z = {};

    for (int j0 = 0; j0 < NTOK; j0 += 32) {
        // ---- K tile (keys x d), shared by both query groups ------------------
        const _Float16* kr0 = Kb + (size_t)(j0 + lo) * 64;
        const _Float16* kr1 = kr0 + 16 * 64;
        v16h ka00 = loadA(kr0, hi);        // key-tile0, d 0..31
        v16h ka01 = loadA(kr0 + 32, hi);   // key-tile0, d 32..63
        v16h ka10 = loadA(kr1, hi);
        v16h ka11 = loadA(kr1 + 32, hi);

        v8f sA0 = wmma16(ka00, qbA0, z); sA0 = wmma16(ka01, qbA1, sA0);
        v8f sB0 = wmma16(ka00, qbB0, z); sB0 = wmma16(ka01, qbB1, sB0);
        v8f sA1 = wmma16(ka10, qbA0, z); sA1 = wmma16(ka11, qbA1, sA1);
        v8f sB1 = wmma16(ka10, qbB0, z); sB1 = wmma16(ka11, qbB1, sB1);

        // ---- P^T = exp2(S^T) (log2e folded into Q); lane-local sums ---------
        float aA = 0.f, aB = 0.f;
        #pragma unroll
        for (int r = 0; r < 8; ++r) {
            sA0[r] = fexp2(sA0[r]); sA1[r] = fexp2(sA1[r]);
            sB0[r] = fexp2(sB0[r]); sB1[r] = fexp2(sB1[r]);
            aA += sA0[r] + sA1[r];
            aB += sB0[r] + sB1[r];
        }
        lsumA += aA; lsumB += aB;

        v16h pbA = buildPB(sA0, sA1, hi);
        v16h pbB = buildPB(sB0, sB1, hi);

        // ---- O^T += V^T P^T : V tiles shared by both query groups -----------
        const _Float16* vr = Vb + (size_t)lo * NTOK + j0;
        #pragma unroll
        for (int t = 0; t < 4; ++t) {
            v16h va = loadA(vr, hi);
            otA[t] = wmma16(va, pbA, otA[t]);
            otB[t] = wmma16(va, pbB, otB[t]);
            vr += (size_t)16 * NTOK;
        }
    }

    // ---- epilogue: combine key-half sums, normalize, store -------------------
    lsumA += xswap16f(lsumA);
    lsumB += xswap16f(lsumB);
    const float invA = 1.0f / lsumA;
    const float invB = 1.0f / lsumB;

    // O^T C layout: lane holds d = t*16 + 8*hi + r for its query column:
    // 8 contiguous floats per tile per lane.
    float* orowA = Ows + (size_t)((b << 12) + i0 + lo) * 64 + 8 * hi;
    float* orowB = orowA + (size_t)16 * 64;
    #pragma unroll
    for (int t = 0; t < 4; ++t) {
        v4f u;
        u = (v4f){otA[t][0], otA[t][1], otA[t][2], otA[t][3]} * invA;
        *(v4f*)(orowA + t * 16) = u;
        u = (v4f){otA[t][4], otA[t][5], otA[t][6], otA[t][7]} * invA;
        *(v4f*)(orowA + t * 16 + 4) = u;
        u = (v4f){otB[t][0], otB[t][1], otB[t][2], otB[t][3]} * invB;
        *(v4f*)(orowB + t * 16) = u;
        u = (v4f){otB[t][4], otB[t][5], otB[t][6], otB[t][7]} * invB;
        *(v4f*)(orowB + t * 16 + 4) = u;
    }
}

// ---------------------------------------------------------------------------
// Kernel 3: out = O @ w_out^T + b_out + xf, written back channels-major.
// ---------------------------------------------------------------------------
__global__ __launch_bounds__(256) void outproj_kernel(
    const float* __restrict__ Ows, const float* __restrict__ wout,
    const float* __restrict__ bout, const float* __restrict__ x,
    float* __restrict__ out)
{
    __shared__ float ov[64][65];
    const int blk   = blockIdx.x;       // 0..127
    const int b     = blk >> 6;
    const int nbase = (blk & 63) << 6;
    const int tid   = threadIdx.x;

    #pragma unroll
    for (int k = 0; k < 16; ++k) {
        int idx = tid + k * 256;
        int d  = idx & 63;
        int nn = idx >> 6;
        ov[nn][d] = Ows[(size_t)((b << 12) + nbase + nn) * 64 + d];
    }
    __syncthreads();

    const int nn = tid & 63;
    const int cb = tid >> 6;
    for (int c = cb; c < 64; c += 4) {
        float acc = bout[c];
        const float* wr = wout + c * 64;    // uniform across the n-group
        #pragma unroll 8
        for (int d = 0; d < 64; ++d) acc += ov[nn][d] * wr[d];
        size_t oi = (size_t)((b << 6) + c) * NTOK + nbase + nn;
        out[oi] = acc + x[oi];              // residual
    }
}

extern "C" void kernel_launch(void* const* d_in, const int* in_sizes, int n_in,
                              void* d_out, int out_size, void* d_ws, size_t ws_size,
                              hipStream_t stream) {
    const float* x     = (const float*)d_in[0];
    const float* wproj = (const float*)d_in[1];
    const float* bproj = (const float*)d_in[2];
    const float* wout  = (const float*)d_in[3];
    const float* bout  = (const float*)d_in[4];
    float* out = (float*)d_out;

    char* ws = (char*)d_ws;
    _Float16* Qh  = (_Float16*)(ws);                    // 1 MB
    _Float16* Kh  = (_Float16*)(ws + (1u << 20));       // 1 MB
    _Float16* Vth = (_Float16*)(ws + (2u << 20));       // 1 MB
    float*    Ows = (float*)   (ws + (3u << 20));       // 2 MB

    qkv_kernel    <<<128, 256, 0, stream>>>(x, wproj, bproj, Qh, Kh, Vth);
    flash_kernel  <<<128,  64, 0, stream>>>(Qh, Kh, Vth, Ows);
    outproj_kernel<<<128, 256, 0, stream>>>(Ows, wout, bout, x, out);
}
